// Baseline_55997783605449
// MI455X (gfx1250) — matrix operand order repaired and checked by reading, not compiled, not run on hardware
//
#include <hip/hip_runtime.h>

#define NN 50000
#define NE 600000
#define FD 128
#define HD 256
#define NL 3
#define NG 256
#define LD 2

typedef __bf16 bf16x16 __attribute__((ext_vector_type(16)));
typedef float  f32x8   __attribute__((ext_vector_type(8)));
typedef float  f32x4   __attribute__((ext_vector_type(4)));
typedef unsigned int u32x4 __attribute__((ext_vector_type(4)));

union Frag16 { bf16x16 v; u32x4 q[2]; };

__device__ __forceinline__ unsigned short f32_to_bf16_rne(float f) {
  unsigned int u = __float_as_uint(f);
  u += 0x7FFFu + ((u >> 16) & 1u);
  return (unsigned short)(u >> 16);
}

// ---------------- WMMA GEMM (K fixed = 128) ----------------
// C[MxN] = A[MxK](bf16,row) * B[KxN], B given as BT[NxK](bf16,row).
// Block: 256 thr = 8 waves -> 128 rows; grid.y picks a 64-wide N slab.
// BT slab (64 x 128 bf16 = 16KB) staged in LDS with 16B row padding.
#define GK 128
#define LDSK (GK + 8)   // padded row stride in halves (272B, multiple of 16B)

__global__ __launch_bounds__(256)
void gemm_bf16_wmma(const unsigned short* __restrict__ A,
                    const unsigned short* __restrict__ BT,
                    float* __restrict__ C,
                    int M, int N) {
  __shared__ unsigned short ldsB[64 * LDSK];
  const int nbase = blockIdx.y * 64;

  // cooperative stage of BT slab: 64 rows x 128 halves, 16B chunks
  {
    const u32x4* __restrict__ g = (const u32x4*)(BT + (size_t)nbase * GK);
#pragma unroll
    for (int i = 0; i < 4; ++i) {               // 4*256 = 1024 chunks
      int t = threadIdx.x + i * 256;
      int row = t >> 4, c = t & 15;             // 16 chunks (128 halves) per row
      *(u32x4*)(ldsB + row * LDSK + c * 8) = g[t];
    }
  }
  __syncthreads();

  const int wave = threadIdx.x >> 5;
  const int lane = threadIdx.x & 31;
  const int mbase = (blockIdx.x * 8 + wave) * 16;
  if (mbase >= M) return;                       // wave-uniform; after the barrier
  const int l  = lane & 15;
  const int hi = lane >> 4;

  f32x8 acc[4] = {};
  const unsigned short* __restrict__ pa = A + (size_t)(mbase + l) * GK + hi * 8;
  const unsigned short* lb = ldsB + (size_t)l * LDSK + hi * 8;

#pragma unroll
  for (int k0 = 0; k0 < GK; k0 += 32) {
    Frag16 a;
    a.q[0] = *(const u32x4*)(pa + k0);          // K = k0+0..7  (hi=0) / +8..15  (hi=1)
    a.q[1] = *(const u32x4*)(pa + k0 + 16);     // K = k0+16..23 (hi=0) / +24..31 (hi=1)
    Frag16 b[4];
#pragma unroll
    for (int nt = 0; nt < 4; ++nt) {            // ds_load_b128 pairs, all before WMMAs
      const unsigned short* pb = lb + nt * 16 * LDSK + k0;
      b[nt].q[0] = *(const u32x4*)(pb);
      b[nt].q[1] = *(const u32x4*)(pb + 16);
    }
#pragma unroll
    for (int nt = 0; nt < 4; ++nt)
      acc[nt] = __builtin_amdgcn_wmma_f32_16x16x32_bf16(
          false, a.v, false, b[nt].v, (short)0, acc[nt], false, false);
  }
#pragma unroll
  for (int nt = 0; nt < 4; ++nt)
#pragma unroll
    for (int i = 0; i < 8; ++i)
      C[(size_t)(mbase + hi * 8 + i) * N + (nbase + nt * 16 + l)] = acc[nt][i];
}

// ---------------- prep kernels ----------------
__global__ void k_deg_init(float* deg) {
  int i = blockIdx.x * blockDim.x + threadIdx.x;
  if (i < NN) deg[i] = 1.0f;                    // self-loop contributes 1
}
__global__ void k_deg_edges(const int* __restrict__ dst, float* deg) {
  int e = blockIdx.x * blockDim.x + threadIdx.x;
  if (e < NE) unsafeAtomicAdd(&deg[dst[e]], 1.0f);
}
__global__ void k_dinv(float* deg) {
  int i = blockIdx.x * blockDim.x + threadIdx.x;
  if (i < NN) deg[i] = rsqrtf(deg[i]);          // deg >= 1 always
}
__global__ void k_prep_weights(const float* __restrict__ Wg, const float* __restrict__ w1,
                               unsigned short* __restrict__ WgT, unsigned short* __restrict__ w1T) {
  int t = blockIdx.x * blockDim.x + threadIdx.x;
  const int wg_total = NL * FD * FD;
  if (t < wg_total) {
    int layer = t / (FD * FD), r = t % (FD * FD);
    int n = r / FD, k = r % FD;
    WgT[(size_t)layer * FD * FD + n * FD + k] = f32_to_bf16_rne(Wg[(size_t)layer * FD * FD + k * FD + n]);
  } else if (t < wg_total + HD * FD) {
    int r = t - wg_total;
    int n = r / FD, k = r % FD;
    w1T[n * FD + k] = f32_to_bf16_rne(w1[k * HD + n]);
  }
}
__global__ void k_cvt_bf16(const float* __restrict__ in, unsigned short* __restrict__ out, int n) {
  int i = blockIdx.x * blockDim.x + threadIdx.x;
  if (i < n) out[i] = f32_to_bf16_rne(in[i]);
}

// ---------------- message passing ----------------
__global__ void k_selfloop_init(const float* __restrict__ h, const float* __restrict__ dinv,
                                float* __restrict__ acc) {
  int t = blockIdx.x * blockDim.x + threadIdx.x;     // NN * 32 threads, 4 floats each
  if (t >= NN * 32) return;
  int n = t >> 5, c = t & 31;
  float s = dinv[n]; s = s * s;
  f32x4 v = ((const f32x4*)h)[(size_t)n * 32 + c];
  v.x *= s; v.y *= s; v.z *= s; v.w *= s;
  ((f32x4*)acc)[(size_t)n * 32 + c] = v;
}
__global__ __launch_bounds__(256)
void k_scatter_edges(const float* __restrict__ h, const float* __restrict__ dinv,
                     const int* __restrict__ src, const int* __restrict__ dst,
                     float* __restrict__ acc) {
  long long t = (long long)blockIdx.x * blockDim.x + threadIdx.x;  // NE*32 threads
  if (t >= (long long)NE * 32) return;
  int e = (int)(t >> 5), c = (int)(t & 31);
  int s = src[e], d = dst[e];
  float nr = dinv[s] * dinv[d];
  f32x4 v = ((const f32x4*)h)[(size_t)s * 32 + c];
  float* p = acc + (size_t)d * FD + c * 4;
  unsafeAtomicAdd(p + 0, v.x * nr);
  unsafeAtomicAdd(p + 1, v.y * nr);
  unsafeAtomicAdd(p + 2, v.z * nr);
  unsafeAtomicAdd(p + 3, v.w * nr);
}
__global__ void k_relu_bias(const float* __restrict__ acc, const float* __restrict__ bgl,
                            float* __restrict__ hf32, unsigned short* __restrict__ xbf) {
  int t = blockIdx.x * blockDim.x + threadIdx.x;
  if (t >= NN * FD) return;
  int d = t & (FD - 1);
  float r = fmaxf(acc[t] + bgl[d], 0.0f);
  hf32[t] = r;
  xbf[t] = f32_to_bf16_rne(r);
}

// ---------------- pooling ----------------
__global__ void k_pool_zero(float* sums, float* counts) {
  int t = blockIdx.x * blockDim.x + threadIdx.x;
  if (t < NG * FD) sums[t] = 0.0f;
  if (t < NG) counts[t] = 0.0f;
}
__global__ void k_pool_scatter(const float* __restrict__ h, const int* __restrict__ batch,
                               float* __restrict__ sums, float* __restrict__ counts) {
  int t = blockIdx.x * blockDim.x + threadIdx.x;     // NN * 32 threads
  if (t >= NN * 32) return;
  int n = t >> 5, c = t & 31;
  int g = batch[n];
  f32x4 v = ((const f32x4*)h)[(size_t)n * 32 + c];
  float* p = sums + (size_t)g * FD + c * 4;
  unsafeAtomicAdd(p + 0, v.x);
  unsafeAtomicAdd(p + 1, v.y);
  unsafeAtomicAdd(p + 2, v.z);
  unsafeAtomicAdd(p + 3, v.w);
  if (c == 0) unsafeAtomicAdd(&counts[g], 1.0f);
}
__global__ void k_pool_div(const float* __restrict__ sums, const float* __restrict__ counts,
                           unsigned short* __restrict__ pooledbf) {
  int t = blockIdx.x * blockDim.x + threadIdx.x;
  if (t >= NG * FD) return;
  int g = t >> 7;
  pooledbf[t] = f32_to_bf16_rne(sums[t] / fmaxf(counts[g], 1.0f));
}

// ---------------- MLP head output ----------------
__global__ void k_mlp_out(const float* __restrict__ mlph, const float* __restrict__ b1,
                          const float* __restrict__ w2, const float* __restrict__ b2,
                          float* __restrict__ out) {
  int t = blockIdx.x * blockDim.x + threadIdx.x;     // NG * LD threads
  if (t >= NG * LD) return;
  int g = t >> 1, j = t & 1;
  float s = b2[j];
  for (int k = 0; k < HD; ++k)
    s += fmaxf(mlph[(size_t)g * HD + k] + b1[k], 0.0f) * w2[k * LD + j];
  out[t] = s;
}

extern "C" void kernel_launch(void* const* d_in, const int* in_sizes, int n_in,
                              void* d_out, int out_size, void* d_ws, size_t ws_size,
                              hipStream_t stream) {
  const float* x  = (const float*)d_in[0];
  const float* Wg = (const float*)d_in[1];
  const float* bg = (const float*)d_in[2];
  const float* w1 = (const float*)d_in[3];
  const float* b1 = (const float*)d_in[4];
  const float* w2 = (const float*)d_in[5];
  const float* b2 = (const float*)d_in[6];
  const int* ei   = (const int*)d_in[7];
  const int* batch= (const int*)d_in[8];
  const int* src = ei;
  const int* dst = ei + NE;
  float* out = (float*)d_out;

  // workspace carve-up (256B aligned)
  size_t off = 0;
  auto carve = [&](size_t bytes) {
    void* p = (char*)d_ws + off;
    off += (bytes + 255) & ~(size_t)255;
    return p;
  };
  float*          dinv    = (float*)carve((size_t)NN * 4);
  unsigned short* xbf     = (unsigned short*)carve((size_t)NN * FD * 2);
  float*          h       = (float*)carve((size_t)NN * FD * 4);
  float*          acc     = (float*)carve((size_t)NN * FD * 4);
  unsigned short* WgT     = (unsigned short*)carve((size_t)NL * FD * FD * 2);
  unsigned short* w1T     = (unsigned short*)carve((size_t)HD * FD * 2);
  float*          sums    = (float*)carve((size_t)NG * FD * 4);
  float*          counts  = (float*)carve((size_t)NG * 4);
  unsigned short* pooled  = (unsigned short*)carve((size_t)NG * FD * 2);
  float*          mlph    = (float*)carve((size_t)NG * HD * 4);

  const int B = 256;
  // degree / normalization
  k_deg_init<<<(NN + B - 1) / B, B, 0, stream>>>(dinv);
  k_deg_edges<<<(NE + B - 1) / B, B, 0, stream>>>(dst, dinv);
  k_dinv<<<(NN + B - 1) / B, B, 0, stream>>>(dinv);
  // weights -> bf16 transposed
  const int wprep = NL * FD * FD + HD * FD;
  k_prep_weights<<<(wprep + B - 1) / B, B, 0, stream>>>(Wg, w1, WgT, w1T);
  // x -> bf16
  k_cvt_bf16<<<((NN * FD) + B - 1) / B, B, 0, stream>>>(x, xbf, NN * FD);

  for (int l = 0; l < NL; ++l) {
    dim3 gg((NN + 127) / 128, FD / 64);
    gemm_bf16_wmma<<<gg, B, 0, stream>>>(xbf, WgT + (size_t)l * FD * FD, h, NN, FD);
    k_selfloop_init<<<(NN * 32 + B - 1) / B, B, 0, stream>>>(h, dinv, acc);
    long long sthreads = (long long)NE * 32;
    k_scatter_edges<<<(unsigned)((sthreads + B - 1) / B), B, 0, stream>>>(h, dinv, src, dst, acc);
    k_relu_bias<<<((NN * FD) + B - 1) / B, B, 0, stream>>>(acc, bg + (size_t)l * FD, h, xbf);
  }

  // mean pool over graphs
  k_pool_zero<<<((NG * FD) + B - 1) / B, B, 0, stream>>>(sums, counts);
  k_pool_scatter<<<(NN * 32 + B - 1) / B, B, 0, stream>>>(h, batch, sums, counts);
  k_pool_div<<<((NG * FD) + B - 1) / B, B, 0, stream>>>(sums, counts, pooled);

  // MLP layer 1 via WMMA: [NG x FD] @ [FD x HD] -> [NG x HD]
  dim3 gm((NG + 127) / 128, HD / 64);
  gemm_bf16_wmma<<<gm, B, 0, stream>>>(pooled, w1T, mlph, NG, HD);
  // ReLU(+b1) and final 2-wide projection
  k_mlp_out<<<((NG * LD) + B - 1) / B, B, 0, stream>>>(mlph, b1, w2, b2, out);
}
